// AttentionBlock_33200097198772
// MI455X (gfx1250) — compile-verified, matrix operand order checked by reading, not run on hardware
//
#include <hip/hip_runtime.h>

#define B_  32
#define C_  512
#define N_  1024
#define G_  32
#define CPG 16          // C_/G_
#define EPS 1e-5f

typedef __attribute__((ext_vector_type(16))) __bf16 v16bf;
typedef __attribute__((ext_vector_type(8)))  float  v8f;

__device__ __forceinline__ v8f wmma_bf16(v16bf a, v16bf b, v8f c) {
    // D = A(16x32 bf16) x B(32x16 bf16) + C(16x16 f32)
    return __builtin_amdgcn_wmma_f32_16x16x32_bf16(
        /*neg_a=*/false, a, /*neg_b=*/false, b,
        /*c_mod=*/(short)0, c, /*reuse_a=*/false, /*reuse_b=*/false);
}

// Core tile compute: A is [M][K] row-major (K contiguous), Bt is [Ncols][K]
// (K contiguous per output column). Wave computes a 32(M) x 64(N) tile.
// Fragment layouts used here (contiguous K order, same for A and B):
//   A 16x32 bf16: lane l holds row (l&15), K half (l>>4), 16 contiguous K vals
//   B 32x16 bf16: lane l holds col (l&15), K half (l>>4), 16 contiguous K vals
__device__ __forceinline__ void gemm_core(const __bf16* __restrict__ A,
                                          const __bf16* __restrict__ Bt,
                                          int K, int m0, int n0,
                                          v8f acc[2][4]) {
    const int lane = threadIdx.x & 31;
    const int lr = lane & 15, lh = lane >> 4;
    const __bf16* Ab = A  + (size_t)(m0 + lr) * K + 16 * lh;
    const __bf16* Bb = Bt + (size_t)(n0 + lr) * K + 16 * lh;
#pragma unroll
    for (int i = 0; i < 2; ++i)
#pragma unroll
        for (int j = 0; j < 4; ++j) acc[i][j] = (v8f){};

    for (int k = 0; k < K; k += 32) {
        v16bf a0 = *(const v16bf*)(Ab + k);
        v16bf a1 = *(const v16bf*)(Ab + (size_t)16 * K + k);
        v16bf b0 = *(const v16bf*)(Bb + k);
        v16bf b1 = *(const v16bf*)(Bb + (size_t)16 * K + k);
        v16bf b2 = *(const v16bf*)(Bb + (size_t)32 * K + k);
        v16bf b3 = *(const v16bf*)(Bb + (size_t)48 * K + k);
        acc[0][0] = wmma_bf16(a0, b0, acc[0][0]);
        acc[0][1] = wmma_bf16(a0, b1, acc[0][1]);
        acc[0][2] = wmma_bf16(a0, b2, acc[0][2]);
        acc[0][3] = wmma_bf16(a0, b3, acc[0][3]);
        acc[1][0] = wmma_bf16(a1, b0, acc[1][0]);
        acc[1][1] = wmma_bf16(a1, b1, acc[1][1]);
        acc[1][2] = wmma_bf16(a1, b2, acc[1][2]);
        acc[1][3] = wmma_bf16(a1, b3, acc[1][3]);
    }
}

// D element (i,j,r): row = m0 + i*16 + r + 8*(lane>>4), col = n0 + j*16 + (lane&15)
#define WAVE_TILE_COORDS()                                  \
    const int wave = threadIdx.x >> 5;                      \
    const int wm = wave & 3, wn = wave >> 2;                \
    const int m0 = blockIdx.x * 128 + wm * 32;              \
    const int n0 = blockIdx.y * 128 + wn * 64;              \
    const int lane = threadIdx.x & 31;                      \
    const int lr = lane & 15, lh = lane >> 4;

// ---------------- weights f32 -> bf16 ----------------
__global__ void k_convert_w(const float* __restrict__ wq, const float* __restrict__ wk,
                            const float* __restrict__ wv, const float* __restrict__ wo,
                            __bf16* q, __bf16* k, __bf16* v, __bf16* o) {
    int i = blockIdx.x * 256 + threadIdx.x;
    if (i < C_ * C_) {
        q[i] = (__bf16)wq[i]; k[i] = (__bf16)wk[i];
        v[i] = (__bf16)wv[i]; o[i] = (__bf16)wo[i];
    }
}

// ---------------- GroupNorm ----------------
// one block per (batch, group); emits f32 xn[b][c][n] (residual) and bf16 ht[b][n][c]
__global__ __launch_bounds__(256)
void k_groupnorm(const float* __restrict__ x, const float* __restrict__ gamma,
                 const float* __restrict__ beta, float* __restrict__ xn,
                 __bf16* __restrict__ ht) {
    const int b = blockIdx.x, g = blockIdx.y, tid = threadIdx.x;
    const float* xp = x + ((size_t)b * C_ + g * CPG) * N_;
    __shared__ float s_sum[256], s_sq[256];
    float sum = 0.f, sq = 0.f;
    for (int i = tid; i < CPG * N_; i += 256) {
        float v = xp[i];
        sum += v; sq += v * v;
    }
    s_sum[tid] = sum; s_sq[tid] = sq;
    __syncthreads();
    for (int s = 128; s > 0; s >>= 1) {
        if (tid < s) { s_sum[tid] += s_sum[tid + s]; s_sq[tid] += s_sq[tid + s]; }
        __syncthreads();
    }
    const float inv_cnt = 1.0f / (CPG * N_);
    const float mean = s_sum[0] * inv_cnt;
    const float var  = s_sq[0] * inv_cnt - mean * mean;
    const float rstd = rsqrtf(var + EPS);
    for (int i = tid; i < CPG * N_; i += 256) {
        const int c = g * CPG + (i >> 10);
        const int n = i & (N_ - 1);
        const float v = (xp[i] - mean) * rstd * gamma[c] + beta[c];
        xn[((size_t)b * C_ + c) * N_ + n] = v;
        ht[((size_t)b * N_ + n) * C_ + c] = (__bf16)v;
    }
}

// ---------------- QKV projection ----------------
// D[o][n] = sum_c W[o][c] * ht[b][n][c];  q,k -> [n][o] (for score), v -> [o][n]
__global__ __launch_bounds__(256)
void k_qkv(const __bf16* __restrict__ wq, const __bf16* __restrict__ wk,
           const __bf16* __restrict__ wv, const float* __restrict__ bq,
           const float* __restrict__ bk, const float* __restrict__ bv,
           const __bf16* __restrict__ ht, __bf16* __restrict__ qt,
           __bf16* __restrict__ kt, __bf16* __restrict__ vmat) {
    const int mode = blockIdx.z % 3;           // 0:q 1:k 2:v
    const int b    = blockIdx.z / 3;
    const __bf16* W    = (mode == 0) ? wq : (mode == 1) ? wk : wv;
    const float*  bias = (mode == 0) ? bq : (mode == 1) ? bk : bv;
    const __bf16* Bt = ht + (size_t)b * N_ * C_;

    WAVE_TILE_COORDS();
    v8f acc[2][4];
    gemm_core(W, Bt, C_, m0, n0, acc);

    if (mode < 2) {
        __bf16* dst = ((mode == 0) ? qt : kt) + (size_t)b * N_ * C_;  // [n][o]
#pragma unroll
        for (int i = 0; i < 2; ++i)
#pragma unroll
            for (int j = 0; j < 4; ++j)
#pragma unroll
                for (int r = 0; r < 8; ++r) {
                    const int o = m0 + i * 16 + r + 8 * lh;
                    const int n = n0 + j * 16 + lr;
                    dst[(size_t)n * C_ + o] = (__bf16)(acc[i][j][r] + bias[o]);
                }
    } else {
        __bf16* dst = vmat + (size_t)b * C_ * N_;  // [o][n]
#pragma unroll
        for (int i = 0; i < 2; ++i)
#pragma unroll
            for (int j = 0; j < 4; ++j)
#pragma unroll
                for (int r = 0; r < 8; ++r) {
                    const int o = m0 + i * 16 + r + 8 * lh;
                    const int n = n0 + j * 16 + lr;
                    dst[(size_t)o * N_ + n] = (__bf16)(acc[i][j][r] + bias[o]);
                }
    }
}

// ---------------- score = (q^T k) * C^-0.5 ----------------
// A = qt[b][n][c], Bt = kt[b][m][c] -> score[b][n][m] f32
__global__ __launch_bounds__(256)
void k_score(const __bf16* __restrict__ qt, const __bf16* __restrict__ kt,
             float* __restrict__ score) {
    const int b = blockIdx.z;
    const __bf16* A  = qt + (size_t)b * N_ * C_;
    const __bf16* Bt = kt + (size_t)b * N_ * C_;
    float* S = score + (size_t)b * N_ * N_;
    const float scale = 0.044194173824159216f;  // 512^-0.5

    WAVE_TILE_COORDS();
    v8f acc[2][4];
    gemm_core(A, Bt, C_, m0, n0, acc);
#pragma unroll
    for (int i = 0; i < 2; ++i)
#pragma unroll
        for (int j = 0; j < 4; ++j)
#pragma unroll
            for (int r = 0; r < 8; ++r) {
                const int n = m0 + i * 16 + r + 8 * lh;
                const int m = n0 + j * 16 + lr;
                S[(size_t)n * N_ + m] = acc[i][j][r] * scale;
            }
}

// ---------------- softmax over rows of score -> bf16 attn ----------------
__global__ __launch_bounds__(256)
void k_softmax(const float* __restrict__ score, __bf16* __restrict__ attn) {
    const size_t row = blockIdx.x;
    const float* s = score + row * N_;
    __bf16* a = attn + row * N_;
    const int tid = threadIdx.x;
    __shared__ float red[256];

    float lv[4];
    float mx = -3.0e38f;
#pragma unroll
    for (int t = 0; t < 4; ++t) { lv[t] = s[tid + t * 256]; mx = fmaxf(mx, lv[t]); }
    red[tid] = mx; __syncthreads();
    for (int st = 128; st > 0; st >>= 1) {
        if (tid < st) red[tid] = fmaxf(red[tid], red[tid + st]);
        __syncthreads();
    }
    mx = red[0]; __syncthreads();

    float sum = 0.f;
#pragma unroll
    for (int t = 0; t < 4; ++t) { lv[t] = __expf(lv[t] - mx); sum += lv[t]; }
    red[tid] = sum; __syncthreads();
    for (int st = 128; st > 0; st >>= 1) {
        if (tid < st) red[tid] += red[tid + st];
        __syncthreads();
    }
    const float inv = 1.0f / red[0];
#pragma unroll
    for (int t = 0; t < 4; ++t) a[tid + t * 256] = (__bf16)(lv[t] * inv);
}

// ---------------- out = attn @ v^T -> outT[b][n][c] ----------------
// A = attn[b][n][m] (K=m), Bt = vmat[b][c][m]
__global__ __launch_bounds__(256)
void k_av(const __bf16* __restrict__ attn, const __bf16* __restrict__ vmat,
          __bf16* __restrict__ outT) {
    const int b = blockIdx.z;
    const __bf16* A  = attn + (size_t)b * N_ * N_;
    const __bf16* Bt = vmat + (size_t)b * C_ * N_;
    __bf16* D = outT + (size_t)b * N_ * C_;

    WAVE_TILE_COORDS();
    v8f acc[2][4];
    gemm_core(A, Bt, N_, m0, n0, acc);
#pragma unroll
    for (int i = 0; i < 2; ++i)
#pragma unroll
        for (int j = 0; j < 4; ++j)
#pragma unroll
            for (int r = 0; r < 8; ++r) {
                const int n = m0 + i * 16 + r + 8 * lh;
                const int c = n0 + j * 16 + lr;
                D[(size_t)n * C_ + c] = (__bf16)acc[i][j][r];
            }
}

// ---------------- final projection + bias + residual ----------------
// y[b][o][n] = sum_c wo[o][c] outT[b][n][c] + bo[o] + xn[b][o][n]
__global__ __launch_bounds__(256)
void k_out(const __bf16* __restrict__ wo, const float* __restrict__ bo,
           const __bf16* __restrict__ outT, const float* __restrict__ xn,
           float* __restrict__ y) {
    const int b = blockIdx.z;
    const __bf16* Bt = outT + (size_t)b * N_ * C_;
    const float* R = xn + (size_t)b * C_ * N_;
    float* Y = y + (size_t)b * C_ * N_;

    WAVE_TILE_COORDS();
    v8f acc[2][4];
    gemm_core(wo, Bt, C_, m0, n0, acc);
#pragma unroll
    for (int i = 0; i < 2; ++i)
#pragma unroll
        for (int j = 0; j < 4; ++j)
#pragma unroll
            for (int r = 0; r < 8; ++r) {
                const int o = m0 + i * 16 + r + 8 * lh;
                const int n = n0 + j * 16 + lr;
                const size_t idx = (size_t)o * N_ + n;
                Y[idx] = acc[i][j][r] + bo[o] + R[idx];
            }
}

extern "C" void kernel_launch(void* const* d_in, const int* in_sizes, int n_in,
                              void* d_out, int out_size, void* d_ws, size_t ws_size,
                              hipStream_t stream) {
    const float* x     = (const float*)d_in[0];
    const float* gamma = (const float*)d_in[1];
    const float* beta  = (const float*)d_in[2];
    const float* wq    = (const float*)d_in[3];
    const float* bq    = (const float*)d_in[4];
    const float* wk    = (const float*)d_in[5];
    const float* bk    = (const float*)d_in[6];
    const float* wv    = (const float*)d_in[7];
    const float* bv    = (const float*)d_in[8];
    const float* wo    = (const float*)d_in[9];
    const float* bo    = (const float*)d_in[10];
    float* y = (float*)d_out;

    // workspace carve (256B aligned)
    char* p = (char*)d_ws;
    auto alloc = [&](size_t bytes) {
        char* r = p;
        p += (bytes + 255) & ~(size_t)255;
        return r;
    };
    float*  xn    = (float*)alloc((size_t)B_ * C_ * N_ * 4);   //  64 MB
    __bf16* ht    = (__bf16*)alloc((size_t)B_ * N_ * C_ * 2);  //  32 MB
    __bf16* qt    = (__bf16*)alloc((size_t)B_ * N_ * C_ * 2);  //  32 MB
    __bf16* kt    = (__bf16*)alloc((size_t)B_ * N_ * C_ * 2);  //  32 MB
    __bf16* vmat  = (__bf16*)alloc((size_t)B_ * C_ * N_ * 2);  //  32 MB
    float*  score = (float*)alloc((size_t)B_ * N_ * N_ * 4);   // 128 MB
    __bf16* attn  = (__bf16*)alloc((size_t)B_ * N_ * N_ * 2);  //  64 MB
    __bf16* outT  = (__bf16*)alloc((size_t)B_ * N_ * C_ * 2);  //  32 MB
    __bf16* wq_b  = (__bf16*)alloc((size_t)C_ * C_ * 2);
    __bf16* wk_b  = (__bf16*)alloc((size_t)C_ * C_ * 2);
    __bf16* wv_b  = (__bf16*)alloc((size_t)C_ * C_ * 2);
    __bf16* wo_b  = (__bf16*)alloc((size_t)C_ * C_ * 2);

    k_convert_w<<<(C_ * C_ + 255) / 256, 256, 0, stream>>>(wq, wk, wv, wo,
                                                           wq_b, wk_b, wv_b, wo_b);
    k_groupnorm<<<dim3(B_, G_), 256, 0, stream>>>(x, gamma, beta, xn, ht);
    k_qkv<<<dim3(C_ / 128, N_ / 128, 3 * B_), 256, 0, stream>>>(
        wq_b, wk_b, wv_b, bq, bk, bv, ht, qt, kt, vmat);
    k_score<<<dim3(N_ / 128, N_ / 128, B_), 256, 0, stream>>>(qt, kt, score);
    k_softmax<<<B_ * N_, 256, 0, stream>>>(score, attn);
    k_av<<<dim3(N_ / 128, C_ / 128, B_), 256, 0, stream>>>(attn, vmat, outT);
    k_out<<<dim3(C_ / 128, N_ / 128, B_), 256, 0, stream>>>(wo_b, bo, outT, xn, y);
}
